// ContrastivePretrainedSAGE_48885317763313
// MI455X (gfx1250) — matrix operand order repaired and checked by reading, not compiled, not run on hardware
//
#include <hip/hip_runtime.h>
#include <hip/hip_bf16.h>

typedef __attribute__((ext_vector_type(16))) __bf16 v16bf;
typedef __attribute__((ext_vector_type(8)))  float  v8f;

#define N_NODES 100000
#define N_EDGES 1600000
#define D_IN    256
#define D_H     128

// ---------------------------------------------------------------------------
// WMMA tile loaders (bf16 A 16x32, bf16 B 32x16), contiguous k order, same for A and B
// A (MxK=16x32): lanes 0-15 -> M=lane, elems 0..15 = K(kk*32 + 0..15)
//                lanes 16-31 -> M=lane-16, elems 0..15 = K(kk*32 + 16..31)
// B (KxN=32x16): lanes 0-15 -> N=lane, elems 0..15 = K(kk*32 + 0..15)
//                lanes 16-31 -> N=lane-16, elems 0..15 = K(kk*32 + 16..31)
// B[k][j] = W[j][k] for y = x @ W.T with W row-major [D_H][D_IN] -> read W rows.
// ---------------------------------------------------------------------------
__device__ __forceinline__ v16bf load_a_x(const float* __restrict__ x,
                                          int row0, int kk, int lane) {
  const int half = lane >> 4, l15 = lane & 15;
  const float* p = x + (size_t)(row0 + l15) * D_IN + kk * 32 + half * 16;
  v16bf a;
#pragma unroll
  for (int i = 0; i < 8; ++i) a[i] = (__bf16)p[i];
#pragma unroll
  for (int i = 0; i < 8; ++i) a[8 + i] = (__bf16)p[8 + i];
  return a;
}

__device__ __forceinline__ v16bf load_b_w(const float* __restrict__ W,
                                          int c0, int kk, int lane) {
  const int half = lane >> 4, l15 = lane & 15;
  const float* p = W + (size_t)(c0 + l15) * D_IN + kk * 32 + half * 16;
  v16bf b;
#pragma unroll
  for (int i = 0; i < 16; ++i) b[i] = (__bf16)p[i];
  return b;
}

// ---------------------------------------------------------------------------
// K1: zero sum/deg/score regions of workspace
// ---------------------------------------------------------------------------
__global__ void __launch_bounds__(256) zero_kernel(float* __restrict__ p, size_t count) {
  size_t i = (size_t)blockIdx.x * blockDim.x + threadIdx.x;
  const size_t stride = (size_t)gridDim.x * blockDim.x;
  for (; i < count; i += stride) p[i] = 0.0f;
}

// ---------------------------------------------------------------------------
// K2: y = x @ W_l.T   [N,256]x[256,128] -> [N,128] f32 (bf16 WMMA, f32 accum)
// one wave per 16x16 output tile; blockIdx = row tile, warp = col tile
// ---------------------------------------------------------------------------
__global__ void __launch_bounds__(256) gemm_y_kernel(const float* __restrict__ x,
                                                     const float* __restrict__ W_l,
                                                     float* __restrict__ y) {
  const int lane = threadIdx.x & 31;
  const int c0   = (threadIdx.x >> 5) * 16;   // 8 col tiles
  const int row0 = blockIdx.x * 16;           // 6250 row tiles
  v8f acc = {};
#pragma unroll
  for (int kk = 0; kk < 8; ++kk) {
    v16bf a = load_a_x(x, row0, kk, lane);
    v16bf b = load_b_w(W_l, c0, kk, lane);
    acc = __builtin_amdgcn_wmma_f32_16x16x32_bf16(false, a, false, b,
                                                  (short)0, acc, false, false);
  }
  const int half = lane >> 4, l15 = lane & 15;
#pragma unroll
  for (int r = 0; r < 8; ++r) {
    const int m = row0 + r + half * 8;        // C layout: VGPR r -> M=r (+8 for hi half)
    y[(size_t)m * D_H + c0 + l15] = acc[r];
  }
}

// ---------------------------------------------------------------------------
// K3: edge scatter: sum[dst] += y[src] (128 f32 atomics / edge), deg[dst] += 1
// one wave per edge; lane holds float4 -> 4 atomics
// ---------------------------------------------------------------------------
__global__ void __launch_bounds__(256) scatter_kernel(const int* __restrict__ ei,
                                                      const float* __restrict__ y,
                                                      float* __restrict__ sum,
                                                      float* __restrict__ deg) {
  const int lane = threadIdx.x & 31;
  const int e = blockIdx.x * 8 + (threadIdx.x >> 5);
  if (e >= N_EDGES) return;
  const int src = ei[e];
  const int dst = ei[N_EDGES + e];
  const float4 v = ((const float4*)(y + (size_t)src * D_H))[lane];
  float* sd = sum + (size_t)dst * D_H + lane * 4;
  atomicAdd(sd + 0, v.x);
  atomicAdd(sd + 1, v.y);
  atomicAdd(sd + 2, v.z);
  atomicAdd(sd + 3, v.w);
  if (lane == 0) atomicAdd(deg + dst, 1.0f);
}

// ---------------------------------------------------------------------------
// K4: fused r = x@W_r.T, s = x@W_res.T, h = relu(sum/deg + b_l + r) + s + b_res,
//     score[m] += sum_j h[m][j] * w_score[j]   (partial over this col tile)
// ---------------------------------------------------------------------------
__global__ void __launch_bounds__(256) fused_kernel(const float* __restrict__ x,
                                                    const float* __restrict__ W_r,
                                                    const float* __restrict__ W_res,
                                                    const float* __restrict__ b_l,
                                                    const float* __restrict__ b_res,
                                                    const float* __restrict__ w_score,
                                                    const float* __restrict__ sum,
                                                    const float* __restrict__ deg,
                                                    float* __restrict__ score) {
  const int lane = threadIdx.x & 31;
  const int c0   = (threadIdx.x >> 5) * 16;
  const int row0 = blockIdx.x * 16;
  v8f accR = {}, accS = {};
#pragma unroll
  for (int kk = 0; kk < 8; ++kk) {
    v16bf a  = load_a_x(x, row0, kk, lane);
    v16bf br = load_b_w(W_r, c0, kk, lane);
    v16bf bs = load_b_w(W_res, c0, kk, lane);
    accR = __builtin_amdgcn_wmma_f32_16x16x32_bf16(false, a, false, br,
                                                   (short)0, accR, false, false);
    accS = __builtin_amdgcn_wmma_f32_16x16x32_bf16(false, a, false, bs,
                                                   (short)0, accS, false, false);
  }
  const int half = lane >> 4, l15 = lane & 15;
  const int j = c0 + l15;
  const float blj = b_l[j], brj = b_res[j], wsj = w_score[j];
#pragma unroll
  for (int r = 0; r < 8; ++r) {
    const int m = row0 + r + half * 8;
    float d = deg[m];
    d = d > 1.0f ? d : 1.0f;
    const float pre = sum[(size_t)m * D_H + j] / d + blj + accR[r];
    const float h = (pre > 0.0f ? pre : 0.0f) + accS[r] + brj;
    float part = h * wsj;
    // reduce over the 16 lanes of this half (j = c0..c0+15); xor masks stay in-half
    part += __shfl_xor(part, 1, 32);
    part += __shfl_xor(part, 2, 32);
    part += __shfl_xor(part, 4, 32);
    part += __shfl_xor(part, 8, 32);
    if (l15 == 0) atomicAdd(score + m, part);   // 8 col tiles contend per node
  }
}

// ---------------------------------------------------------------------------
// K5: out = sigmoid(alpha)*reranker + (1-sigmoid(alpha))*(score + b_score)
// ---------------------------------------------------------------------------
__global__ void __launch_bounds__(256) blend_kernel(const float* __restrict__ rr,
                                                    const float* __restrict__ score,
                                                    const float* __restrict__ b_score,
                                                    const float* __restrict__ alpha,
                                                    float* __restrict__ out) {
  const int n = blockIdx.x * blockDim.x + threadIdx.x;
  if (n >= N_NODES) return;
  const float a = 1.0f / (1.0f + __expf(-alpha[0]));
  out[n] = a * rr[n] + (1.0f - a) * (score[n] + b_score[0]);
}

extern "C" void kernel_launch(void* const* d_in, const int* in_sizes, int n_in,
                              void* d_out, int out_size, void* d_ws, size_t ws_size,
                              hipStream_t stream) {
  const float* x       = (const float*)d_in[0];
  const int*   ei      = (const int*)d_in[1];
  const float* rr      = (const float*)d_in[2];
  const float* W_l     = (const float*)d_in[3];
  const float* b_l     = (const float*)d_in[4];
  const float* W_r     = (const float*)d_in[5];
  const float* W_res   = (const float*)d_in[6];
  const float* b_res   = (const float*)d_in[7];
  const float* w_score = (const float*)d_in[8];
  const float* b_score = (const float*)d_in[9];
  const float* alpha   = (const float*)d_in[10];
  float* out = (float*)d_out;

  // workspace layout (floats): y[N*128] | sum[N*128] | deg[N] | score[N]
  float* ws    = (float*)d_ws;
  float* y     = ws;
  float* sum   = ws + (size_t)N_NODES * D_H;
  float* deg   = sum + (size_t)N_NODES * D_H;
  float* score = deg + N_NODES;

  const size_t zero_count = (size_t)N_NODES * D_H + 2 * (size_t)N_NODES;
  zero_kernel<<<2048, 256, 0, stream>>>(sum, zero_count);

  gemm_y_kernel<<<N_NODES / 16, 256, 0, stream>>>(x, W_l, y);

  scatter_kernel<<<N_EDGES / 8, 256, 0, stream>>>(ei, y, sum, deg);

  fused_kernel<<<N_NODES / 16, 256, 0, stream>>>(x, W_r, W_res, b_l, b_res,
                                                 w_score, sum, deg, score);

  blend_kernel<<<(N_NODES + 255) / 256, 256, 0, stream>>>(rr, score, b_score,
                                                          alpha, out);
}
